// SlayerNet_86079734546561
// MI455X (gfx1250) — matrix operand order repaired and checked by reading, not compiled, not run on hardware
//
#include <hip/hip_runtime.h>
#include <hip/hip_bf16.h>

// ---------------- problem dims (match reference) ----------------
#define BSZ   256
#define TLEN  500
#define ENC   700
#define KPAD  704                   // ENC padded to a multiple of 32
#define HID   128
#define MROWS (BSZ * TLEN)          // 128000 GEMM rows, divisible by 16
#define NCHUNK (KPAD / 32)          // 22 K-chunks

// neuron constants: tau = 20, theta = 1, scaleRef = 1
#define DDEC  0.9512294245007140f   // exp(-1/20)
#define CALP  0.1359140914229523f   // e/20        (alpha filter mult=1)
#define CREF  (-0.2718281828459045f)// -2*e/20     (refractory kernel)

typedef __attribute__((ext_vector_type(16))) _Float16 v16h;
typedef __attribute__((ext_vector_type(8)))  _Float16 v8h;
typedef __attribute__((ext_vector_type(8)))  float    v8f;

// LDS pitch for a 32x128 w1 chunk, in halfs. 40 halfs = 80 B = 20 dwords:
// bank stride 20 across the 16 lanes of a fragment read -> conflict-free,
// 16-byte aligned so fragment reads lower to ds_load_b128.
#define BPITCH   40
#define BUFBYTES (HID * BPITCH * 2)     // 10240 B per buffer

// ---------------------------------------------------------------------------
// Kernel 0: one-time w1 f32 -> f16 with K padded to 704 (zero fill).
// Keeps the GEMM hot loop free of conversion and B-side tail logic.
// ---------------------------------------------------------------------------
__global__ __launch_bounds__(256)
void w1_to_f16(const float* __restrict__ w1, _Float16* __restrict__ W1h) {
    const int idx = blockIdx.x * 256 + threadIdx.x;
    if (idx >= HID * KPAD) return;
    const int h = idx / KPAD;
    const int c = idx % KPAD;
    W1h[idx] = (_Float16)((c < ENC) ? w1[h * ENC + c] : 0.f);
}

// ---------------------------------------------------------------------------
// Kernel 1: H[m, h] = sum_c x[m, c] * w1[h, c],  m = b*T + t  (f32 in, f16 out)
// 8 waves / block, each wave: one 16-row M tile x all 8 N tiles.
// B chunks staged by gfx1250 async DMA (global_load_async_to_lds_b128) into a
// double-buffered LDS region; A fragments software-pipelined one chunk ahead.
// ---------------------------------------------------------------------------
__global__ __launch_bounds__(256)
void slayer_gemm_wmma(const float* __restrict__ x,
                      const _Float16* __restrict__ W1h,
                      _Float16* __restrict__ H) {
    __shared__ __align__(16) _Float16 Blds[2][HID * BPITCH];

    const int tid  = threadIdx.x;
    const int wv   = tid >> 5;          // wave 0..7
    const int lane = tid & 31;
    const int mlo  = lane & 15;         // row within tile / N within tile
    const int hi   = lane >> 4;         // half-wave select
    const size_t m0 = ((size_t)blockIdx.x * 8 + wv) * 16;
    const float* xrow = x + (m0 + mlo) * (size_t)ENC;

    // ---- per-thread async-DMA segment descriptors (2 x 16B per thread) ----
    // 512 16-byte segments cover the 32x128 f16 chunk; lane-private LDS dest
    // addresses implement the BPITCH padding for free.
    const int s0 = tid, s1 = tid + 256;
    const int h0 = s0 >> 2, p0 = s0 & 3;
    const int h1 = s1 >> 2, p1 = s1 & 3;
    const uint32_t ldsbase = (uint32_t)(uintptr_t)(&Blds[0][0]);
    const uint32_t loff0 = (uint32_t)(h0 * (BPITCH * 2) + p0 * 16);
    const uint32_t loff1 = (uint32_t)(h1 * (BPITCH * 2) + p1 * 16);

    auto issue_chunk = [&](int buf, int kc) {
        const _Float16* g0 = W1h + (size_t)h0 * KPAD + kc + p0 * 8;
        const _Float16* g1 = W1h + (size_t)h1 * KPAD + kc + p1 * 8;
        const uint32_t l0 = ldsbase + (uint32_t)buf * BUFBYTES + loff0;
        const uint32_t l1 = ldsbase + (uint32_t)buf * BUFBYTES + loff1;
        asm volatile("global_load_async_to_lds_b128 %0, %1, off"
                     :: "v"(l0), "v"(g0) : "memory");
        asm volatile("global_load_async_to_lds_b128 %0, %1, off"
                     :: "v"(l1), "v"(g1) : "memory");
    };

    // ---- A fragment (16x32 f16) loader, ISA lane/K mapping:
    // lanes 0-15: a[0..7]=K kc..kc+7, a[8..15]=K kc+16..kc+23; lanes 16-31: +8
    auto loadA = [&](int kc) -> v16h {
        const int kbase = kc + hi * 8;
        v16h a;
        if (kc + 32 <= ENC) {
            const float4* q0 = reinterpret_cast<const float4*>(xrow + kbase);
            const float4* q1 = reinterpret_cast<const float4*>(xrow + kbase + 16);
            float4 f0 = q0[0], f1 = q0[1], f2 = q1[0], f3 = q1[1];
            a[0]=(_Float16)f0.x;  a[1]=(_Float16)f0.y;  a[2]=(_Float16)f0.z;  a[3]=(_Float16)f0.w;
            a[4]=(_Float16)f1.x;  a[5]=(_Float16)f1.y;  a[6]=(_Float16)f1.z;  a[7]=(_Float16)f1.w;
            a[8]=(_Float16)f2.x;  a[9]=(_Float16)f2.y;  a[10]=(_Float16)f2.z; a[11]=(_Float16)f2.w;
            a[12]=(_Float16)f3.x; a[13]=(_Float16)f3.y; a[14]=(_Float16)f3.z; a[15]=(_Float16)f3.w;
        } else {                        // zero-padded K tail (700 -> 704)
#pragma unroll
            for (int i = 0; i < 16; ++i) {
                const int k = kbase + i + (i & 8);
                a[i] = (_Float16)((k < ENC) ? xrow[k] : 0.f);
            }
        }
        return a;
    };

    v8f acc[8] = {};                    // 8 N-tiles of 16x16 f32

    issue_chunk(0, 0);                  // prime the pipeline
    v16h a = loadA(0);

    for (int i = 0; i < NCHUNK; ++i) {
        // my async writes for chunk i are done; barrier makes everyone's
        // visible and also retires all reads of the buffer we refill next.
        asm volatile("s_wait_asynccnt 0x0" ::: "memory");
        __syncthreads();

        if (i + 1 < NCHUNK) issue_chunk((i + 1) & 1, (i + 1) * 32);

        const v16h acur = a;
        if (i + 1 < NCHUNK) a = loadA((i + 1) * 32);   // prefetch under WMMAs

        const _Float16* bb = Blds[i & 1];
        v16h bf[8];
#pragma unroll
        for (int n0 = 0; n0 < 8; ++n0) {               // burst the ds loads
            const int off = (n0 * 16 + mlo) * BPITCH + hi * 8;
            v8h blo = *reinterpret_cast<const v8h*>(&bb[off]);
            v8h bhi = *reinterpret_cast<const v8h*>(&bb[off + 16]);
#pragma unroll
            for (int e = 0; e < 8; ++e) { bf[n0][e] = blo[e]; bf[n0][e + 8] = bhi[e]; }
        }
#pragma unroll
        for (int n0 = 0; n0 < 8; ++n0) {
            acc[n0] = __builtin_amdgcn_wmma_f32_16x16x32_f16(
                /*neg_a=*/false, acur, /*neg_b=*/false, bf[n0],
                /*c_mod=*/(short)0, acc[n0],
                /*reuse_a=*/false, /*reuse_b=*/false);
        }
    }

    // ---- store H as f16 (halves intermediate traffic vs f32) ----
#pragma unroll
    for (int n0 = 0; n0 < 8; ++n0) {
#pragma unroll
        for (int r = 0; r < 8; ++r) {
            const size_t m = m0 + r + 8 * hi;       // C/D layout: M = r + 8*(lane/16)
            const int    n = n0 * 16 + mlo;         // N = lane % 16
            H[m * HID + n] = (_Float16)acc[n0][r];
        }
    }
}

// ---------------------------------------------------------------------------
// Kernel 2: fused temporal recurrences (alpha PSP + spike/refractory) for
// layer 1, 128-wide weighted reduction, and layer-2 recurrences on thread 0.
// One block per batch element, thread = hidden unit.
// ---------------------------------------------------------------------------
__global__ __launch_bounds__(128)
void slayer_recurrence(const _Float16* __restrict__ H,
                       const float* __restrict__ w2,
                       float* __restrict__ out) {
    __shared__ float red[4];

    const int b    = blockIdx.x;
    const int h    = threadIdx.x;       // 0..127
    const int wv   = h >> 5;
    const int lane = h & 31;
    const float w2f = w2[h];

    // layer-1 per-(b,h) state
    float pa = 0.f, qa = 0.f;           // alpha filter (PSP)
    float pr = 0.f, qr = 0.f;           // refractory accumulator
    // layer-2 state (live on thread 0 only)
    float pa2 = 0.f, qa2 = 0.f, pr2 = 0.f, qr2 = 0.f;

    const _Float16* Hb = H + ((size_t)b * TLEN) * HID + h;

    for (int t = 0; t < TLEN; ++t) {
        const float hraw = (float)Hb[(size_t)t * HID];

        // alpha filter: output at t uses only past inputs (h[0] = 0)
        qa = DDEC * qa + (CALP * DDEC) * pa;
        pa = DDEC * pa + hraw;
        // refractory response of past spikes, then threshold
        qr = DDEC * qr + (CREF * DDEC) * pr;
        const float u = qa + qr;
        const float s = (u >= 1.0f) ? 1.0f : 0.0f;
        pr = DDEC * pr + s;

        // o_t = sum_h s1 * w2[h] : wave32 shuffle reduce + LDS combine
        float c = s * w2f;
#pragma unroll
        for (int off = 16; off >= 1; off >>= 1)
            c += __shfl_xor(c, off, 32);
        if (lane == 0) red[wv] = c;
        __syncthreads();

        if (threadIdx.x == 0) {
            const float o = red[0] + red[1] + red[2] + red[3];
            // layer-2 alpha filter then spike
            qa2 = DDEC * qa2 + (CALP * DDEC) * pa2;
            pa2 = DDEC * pa2 + o;
            qr2 = DDEC * qr2 + (CREF * DDEC) * pr2;
            const float u2 = qa2 + qr2;
            const float s2 = (u2 >= 1.0f) ? 1.0f : 0.0f;
            pr2 = DDEC * pr2 + s2;
            out[(size_t)b * TLEN + t] = s2;     // (B, T, 1) layout
        }
        __syncthreads();                // red[] reuse next t
    }
}

// ---------------------------------------------------------------------------
extern "C" void kernel_launch(void* const* d_in, const int* in_sizes, int n_in,
                              void* d_out, int out_size, void* d_ws, size_t ws_size,
                              hipStream_t stream) {
    const float* x  = (const float*)d_in[0];   // (B, T, ENC) f32
    const float* w1 = (const float*)d_in[1];   // (HID, ENC)  f32
    const float* w2 = (const float*)d_in[2];   // (1, HID)    f32
    float* out = (float*)d_out;                // (B, T, 1)   f32

    // workspace layout: [ W1h f16 (128 x 704) | H f16 (128000 x 128) ]
    _Float16* W1h = (_Float16*)d_ws;                              // 180224 B
    _Float16* H   = (_Float16*)((char*)d_ws + HID * KPAD * 2);    // 32.8 MB

    w1_to_f16<<<(HID * KPAD + 255) / 256, 256, 0, stream>>>(w1, W1h);
    // GEMM: 128000 rows / (16 rows/wave * 8 waves/block) = 1000 blocks
    slayer_gemm_wmma<<<MROWS / 128, 256, 0, stream>>>(x, W1h, H);
    // Recurrences: one block per batch element
    slayer_recurrence<<<BSZ, 128, 0, stream>>>(H, w2, out);
}
